// model2_9895604650534
// MI455X (gfx1250) — matrix operand order repaired and checked by reading, not compiled, not run on hardware
//
#include <hip/hip_runtime.h>

#define G_    16
#define NPG_  8192
#define NN_   (G_*NPG_)          // 131072 nodes
#define EE_   1048576            // edges
#define NX_   22
#define NY_   22
#define CC_   (G_*NX_*NY_)       // 7744 pooled voxel-nodes
#define EPS_  1e-5f
#define SX_   (16.0f/346.0f)
#define SY_   (12.0f/260.0f)
#define APAD  40                 // padded LDS row (halves): 80B = 20 banks, conflict-free b128

typedef __attribute__((ext_vector_type(16))) _Float16 v16h;
typedef __attribute__((ext_vector_type(8)))  _Float16 v8h;
typedef __attribute__((ext_vector_type(8)))  float    v8f;

// ---- order-preserving float <-> uint keys for atomicMax-based segment_max ----
__device__ __forceinline__ unsigned f2key(float f){
  unsigned u = __float_as_uint(f);
  return (u & 0x80000000u) ? ~u : (u | 0x80000000u);
}
__device__ __forceinline__ float key2f(unsigned k){
  unsigned u = (k & 0x80000000u) ? (k & 0x7FFFFFFFu) : ~k;
  return __uint_as_float(u);
}
__device__ __forceinline__ float eluf(float x){ return x > 0.f ? x : expm1f(x); }

// ---------------------------------------------------------------- utilities
__global__ void k_zero(float* __restrict__ p, long long n){
  long long i = (long long)blockIdx.x*blockDim.x + threadIdx.x;
  if (i < n) p[i] = 0.f;
}

// Wcat[ci][k*Cout+c] = W[k,ci,c], zero-padded to 32 input rows, f16
__global__ void k_prep_w(const float* __restrict__ W, _Float16* __restrict__ Wh,
                         int Cin, int Cout){
  int ncol = 8*Cout;
  int i = blockIdx.x*blockDim.x + threadIdx.x;
  if (i >= 32*ncol) return;
  int ci = i / ncol, col = i % ncol;
  int k = col / Cout, c = col % Cout;
  float v = (ci < Cin) ? W[(size_t)(k*Cin + ci)*Cout + c] : 0.f;
  Wh[i] = (_Float16)v;
}

// X[n,Cin] f32 -> Xh[n,32] f16 zero-padded
__global__ void k_pack_x(const float* __restrict__ X, _Float16* __restrict__ Xh,
                         int n, int Cin){
  long long i = (long long)blockIdx.x*blockDim.x + threadIdx.x;
  if (i >= (long long)n*32) return;
  int node = (int)(i >> 5), c = (int)(i & 31);
  Xh[i] = (c < Cin) ? (_Float16)X[(size_t)node*Cin + c] : (_Float16)0.f;
}

// ------------------------- WMMA GEMM: Y[M,Ncols] = Xh[M,32] @ Wh[32,Ncols] ---
// Block: 64 rows x min(128,Ncols) cols. 8 waves, each owns one 16-col tile and
// iterates 4 row tiles, reusing its B fragment -> 4 back-to-back WMMAs/wave.
// LDS tiles padded to 40-half rows so fragment reads are conflict-free b128.
__global__ void __launch_bounds__(256)
k_gemm_wmma(const _Float16* __restrict__ Xh, const _Float16* __restrict__ Wh,
            _Float16* __restrict__ Y, int M, int Ncols){
  (void)M;
  __shared__ __align__(16) _Float16 sA[64*APAD];
  __shared__ __align__(16) _Float16 sBt[256*APAD];   // transposed: [col][k]
  int tid = threadIdx.x;
  int row0 = blockIdx.x << 6;
  // A tile: 64x32 halves, vector (b32) global reads, padded LDS rows
  const unsigned* Xu = (const unsigned*)(Xh + (size_t)row0*32);
  for (int i = tid; i < 64*16; i += 256){
    int r = i >> 4, c = i & 15;
    *(unsigned*)&sA[r*APAD + 2*c] = Xu[r*16 + c];
  }
  // B tile: transpose Wh[32,Ncols] -> sBt[Ncols][32] (tiny, L1-resident)
  for (int i = tid; i < Ncols*16; i += 256){
    int c = i >> 4, kp = (i & 15)*2;
    sBt[c*APAD + kp]     = Wh[(size_t)kp*Ncols + c];
    sBt[c*APAD + kp + 1] = Wh[(size_t)(kp+1)*Ncols + c];
  }
  __syncthreads();
  int wave = tid >> 5, lane = tid & 31;
  int col0 = (blockIdx.y*8 + wave) << 4;
  if (col0 >= Ncols) return;              // wave-uniform: EXEC stays all-ones
  int half = lane >> 4;                   // 0: lanes 0-15, 1: lanes 16-31
  int l15  = lane & 15;
  // B fragment in this code's contiguous k order (A uses the same): lanes<16 hold K=0..15 of col=lane; lanes>=16 K=16..31
  int kOffB = half ? 16 : 0;
  const _Float16* bp = &sBt[(col0 + l15)*APAD + kOffB];
  v8h b0 = *(const v8h*)bp;
  v8h b1 = *(const v8h*)(bp + 8);
  v16h b = __builtin_shufflevector(b0, b1, 0,1,2,3,4,5,6,7,8,9,10,11,12,13,14,15);
  // A fragment in this code's contiguous k order (matches B): lanes<16 hold K 0..15; lanes>=16 K 16..31
  int kOffA = half ? 16 : 0;
  int rbase = half ? 8 : 0;
  #pragma unroll
  for (int rt = 0; rt < 4; ++rt){
    const _Float16* ap = &sA[(rt*16 + l15)*APAD + kOffA];
    v8h a0 = *(const v8h*)ap;
    v8h a1 = *(const v8h*)(ap + 8);
    v16h a = __builtin_shufflevector(a0, a1, 0,1,2,3,4,5,6,7,8,9,10,11,12,13,14,15);
    v8f acc = {};
    acc = __builtin_amdgcn_wmma_f32_16x16x32_f16(
        /*neg_a=*/false, a, /*neg_b=*/false, b,
        /*c_mod=*/(short)0, acc, /*reuse_a=*/false, /*reuse_b=*/false);
    // D layout: element v -> row v + (lane<16?0:8), col = lane&15
    int rb = row0 + rt*16 + rbase;
    #pragma unroll
    for (int v = 0; v < 8; ++v)
      Y[(size_t)(rb + v)*Ncols + col0 + l15] = (_Float16)acc[v];
  }
}

// degree-1 open B-spline basis: 8 trilinear weights per edge
__global__ void k_basis(const float* __restrict__ attr, float* __restrict__ B, int E){
  int e = blockIdx.x*blockDim.x + threadIdx.x;
  if (e >= E) return;
  float u0 = attr[(size_t)e*3], u1 = attr[(size_t)e*3+1], u2 = attr[(size_t)e*3+2];
  float w0[2] = {1.f-u0, u0}, w1[2] = {1.f-u1, u1}, w2[2] = {1.f-u2, u2};
  #pragma unroll
  for (int k = 0; k < 8; ++k)
    B[(size_t)e*8 + k] = w0[(k>>2)&1]*w1[(k>>1)&1]*w2[k&1];
}

__global__ void k_count(const int* __restrict__ dst, float* __restrict__ cnt, int E){
  int e = blockIdx.x*blockDim.x + threadIdx.x;
  if (e >= E) return;
  atomicAdd(&cnt[dst[e]], 1.f);
}

// per-(edge,channel): msg = sum_k B[e,k] * Y[src, k*Cout+c]; scatter-add to dst
__global__ void k_edge_msg(const _Float16* __restrict__ Y, const float* __restrict__ B,
                           const int* __restrict__ src, const int* __restrict__ dst,
                           float* __restrict__ agg, int E, int Cout){
  long long gid = (long long)blockIdx.x*blockDim.x + threadIdx.x;
  if (gid >= (long long)E*Cout) return;
  int c = (int)(gid % Cout);
  int e = (int)(gid / Cout);
  int s = src[e], d = dst[e];
  const _Float16* y = Y + (size_t)s*(8*Cout) + c;
  const float* be = B + (size_t)e*8;
  float m = 0.f;
  #pragma unroll
  for (int k = 0; k < 8; ++k) m += be[k]*(float)y[k*Cout];
  atomicAdd(&agg[(size_t)d*Cout + c], m);
}

// pooled-graph variant: endpoints are cluster ids, self-loops masked out
__global__ void k_edge_msg_pool(const _Float16* __restrict__ Y, const float* __restrict__ B,
                                const int* __restrict__ src, const int* __restrict__ dst,
                                const int* __restrict__ cl,
                                float* __restrict__ agg, int E, int Cout){
  long long gid = (long long)blockIdx.x*blockDim.x + threadIdx.x;
  if (gid >= (long long)E*Cout) return;
  int c = (int)(gid % Cout);
  int e = (int)(gid / Cout);
  int ns = cl[src[e]], nd = cl[dst[e]];
  if (ns == nd) return;                    // remove_self_loops mask
  const _Float16* y = Y + (size_t)ns*(8*Cout) + c;
  const float* be = B + (size_t)e*8;
  float m = 0.f;
  #pragma unroll
  for (int k = 0; k < 8; ++k) m += be[k]*(float)y[k*Cout];
  atomicAdd(&agg[(size_t)nd*Cout + c], m);
}

// t = elu(agg / max(cnt,1)) in place; accumulate weighted sum/sumsq/count
__global__ void k_elu_stats(float* __restrict__ t, const float* __restrict__ cnt,
                            const float* __restrict__ nmask, float* __restrict__ stats,
                            int n, int Cc){
  __shared__ float sred[65];
  int tid = threadIdx.x;
  long long gid = (long long)blockIdx.x*blockDim.x + tid;
  if (tid < 2*Cc+1) sred[tid] = 0.f;
  __syncthreads();
  if (gid < (long long)n*Cc){
    int c = (int)(gid % Cc);
    int node = (int)(gid / Cc);
    float v = t[gid] / fmaxf(cnt[node], 1.f);
    v = eluf(v);
    t[gid] = v;
    float w = nmask ? nmask[node] : 1.f;
    atomicAdd(&sred[c], v*w);
    atomicAdd(&sred[Cc+c], v*v*w);
    if (c == 0) atomicAdd(&sred[2*Cc], w);
  }
  __syncthreads();
  if (tid < 2*Cc+1) atomicAdd(&stats[tid], sred[tid]);
}

// training-mode BN (biased variance) + optional residual add
__global__ void k_bn(const float* __restrict__ t, const float* __restrict__ stats,
                     const float* __restrict__ g, const float* __restrict__ b,
                     const float* __restrict__ resid, float* __restrict__ out,
                     int n, int Cc){
  long long gid = (long long)blockIdx.x*blockDim.x + threadIdx.x;
  if (gid >= (long long)n*Cc) return;
  int c = (int)(gid % Cc);
  float nn = fmaxf(stats[2*Cc], 1.f);
  float m = stats[c]/nn;
  float v = stats[Cc+c]/nn - m*m;
  float y = (t[gid]-m)*rsqrtf(v + EPS_)*g[c] + b[c];
  out[gid] = resid ? (y + resid[gid]) : y;
}

// ---------------------------------------------------------------- voxel pool
__global__ void k_pool_scatter(const float* __restrict__ x, const float* __restrict__ pos,
                               const int* __restrict__ batch, int* __restrict__ cluster,
                               float* __restrict__ cntc, float* __restrict__ psum,
                               unsigned* __restrict__ xkey, int n){
  int i = blockIdx.x*blockDim.x + threadIdx.x;
  if (i >= n) return;
  float px = pos[(size_t)i*3], py = pos[(size_t)i*3+1], pz = pos[(size_t)i*3+2];
  int ix = min(max((int)floorf(px/SX_), 0), NX_-1);
  int iy = min(max((int)floorf(py/SY_), 0), NY_-1);
  int cl = batch[i]*(NX_*NY_) + iy*NX_ + ix;
  cluster[i] = cl;
  atomicAdd(&cntc[cl], 1.f);
  atomicAdd(&psum[(size_t)cl*3+0], px);
  atomicAdd(&psum[(size_t)cl*3+1], py);
  atomicAdd(&psum[(size_t)cl*3+2], pz);
  #pragma unroll
  for (int c = 0; c < 32; ++c)
    atomicMax(&xkey[(size_t)cl*32 + c], f2key(x[(size_t)i*32 + c]));
}

__global__ void k_pool_finalize(const float* __restrict__ cntc, const float* __restrict__ psum,
                                const unsigned* __restrict__ xkey, float* __restrict__ newx,
                                float* __restrict__ npos, float* __restrict__ vmask,
                                unsigned* __restrict__ mxkey, int C){
  int i = blockIdx.x*blockDim.x + threadIdx.x;
  if (i >= C) return;
  if (i == 0) *mxkey = f2key(0.0f);          // seed global |cart| max
  float cnt = cntc[i];
  bool valid = cnt > 0.f;
  vmask[i] = valid ? 1.f : 0.f;
  float ic = 1.f / fmaxf(cnt, 1.f);
  npos[(size_t)i*3+0] = psum[(size_t)i*3+0]*ic;
  npos[(size_t)i*3+1] = psum[(size_t)i*3+1]*ic;
  npos[(size_t)i*3+2] = psum[(size_t)i*3+2]*ic;
  #pragma unroll
  for (int c = 0; c < 32; ++c)
    newx[(size_t)i*32 + c] = valid ? key2f(xkey[(size_t)i*32 + c]) : 0.f;
}

__global__ void k_pool_edge1(const int* __restrict__ src, const int* __restrict__ dst,
                             const int* __restrict__ cl, const float* __restrict__ npos,
                             float* __restrict__ cart, unsigned* __restrict__ mxkey, int E){
  __shared__ unsigned smax;
  if (threadIdx.x == 0) smax = 0u;
  __syncthreads();
  int e = blockIdx.x*blockDim.x + threadIdx.x;
  if (e < E){
    int ns = cl[src[e]], nd = cl[dst[e]];
    float c0 = npos[(size_t)nd*3+0] - npos[(size_t)ns*3+0];
    float c1 = npos[(size_t)nd*3+1] - npos[(size_t)ns*3+1];
    float c2 = npos[(size_t)nd*3+2] - npos[(size_t)ns*3+2];
    cart[(size_t)e*3+0] = c0; cart[(size_t)e*3+1] = c1; cart[(size_t)e*3+2] = c2;
    if (ns != nd){
      float m = fmaxf(fmaxf(fabsf(c0), fabsf(c1)), fabsf(c2));
      atomicMax(&smax, f2key(m));
    }
  }
  __syncthreads();
  if (threadIdx.x == 0) atomicMax(mxkey, smax);
}

__global__ void k_pool_attr_basis(const float* __restrict__ cart, const unsigned* __restrict__ mxkey,
                                  const int* __restrict__ src, const int* __restrict__ dst,
                                  const int* __restrict__ cl, float* __restrict__ B,
                                  float* __restrict__ cnt2, int E){
  int e = blockIdx.x*blockDim.x + threadIdx.x;
  if (e >= E) return;
  float mx = fmaxf(key2f(*mxkey), 1e-12f);
  float inv = 0.5f/mx;
  float u[3];
  #pragma unroll
  for (int d = 0; d < 3; ++d)
    u[d] = fminf(fmaxf(cart[(size_t)e*3+d]*inv + 0.5f, 0.f), 1.f);
  float w0[2] = {1.f-u[0], u[0]}, w1[2] = {1.f-u[1], u[1]}, w2[2] = {1.f-u[2], u[2]};
  #pragma unroll
  for (int k = 0; k < 8; ++k)
    B[(size_t)e*8 + k] = w0[(k>>2)&1]*w1[(k>>1)&1]*w2[k&1];
  int ns = cl[src[e]], nd = cl[dst[e]];
  if (ns != nd) atomicAdd(&cnt2[nd], 1.f);
}

// ---------------------------------------------------------------- 4x4 pool + FC
__global__ void k_maxpool16(const float* __restrict__ x, const float* __restrict__ npos,
                            const float* __restrict__ vmask, unsigned* __restrict__ key16,
                            float* __restrict__ cnt16, int C){
  int i = blockIdx.x*blockDim.x + threadIdx.x;
  if (i >= C) return;
  if (vmask[i] == 0.f) return;
  int ix = min(max((int)floorf(npos[(size_t)i*3+0]/0.25f), 0), 3);
  int iy = min(max((int)floorf(npos[(size_t)i*3+1]/0.25f), 0), 3);
  int g = i / (NX_*NY_);
  int c16 = g*16 + iy*4 + ix;
  #pragma unroll
  for (int c = 0; c < 32; ++c)
    atomicMax(&key16[(size_t)c16*32 + c], f2key(x[(size_t)i*32 + c]));
  atomicAdd(&cnt16[c16], 1.f);
}

__global__ void k_feat(const unsigned* __restrict__ key16, const float* __restrict__ cnt16,
                       float* __restrict__ feat){
  int i = blockIdx.x*blockDim.x + threadIdx.x;   // over 256*32
  if (i >= G_*16*32) return;
  int c16 = i >> 5, c = i & 31;
  int g = c16 >> 4, q = c16 & 15;
  feat[(size_t)g*512 + q*32 + c] = (cnt16[c16] > 0.f) ? key2f(key16[i]) : 0.f;
}

__global__ void k_fc(const float* __restrict__ feat, const float* __restrict__ Wfc,
                     float* __restrict__ out){
  int i = threadIdx.x;                            // 32 threads: G*2 outputs
  if (i >= G_*2) return;
  int g = i >> 1, o = i & 1;
  float s = 0.f;
  for (int j = 0; j < 512; ++j) s += feat[(size_t)g*512 + j]*Wfc[(size_t)j*2 + o];
  out[(size_t)g*2 + o] = s;
}

// ================================================================ launcher
extern "C" void kernel_launch(void* const* d_in, const int* in_sizes, int n_in,
                              void* d_out, int out_size, void* d_ws, size_t ws_size,
                              hipStream_t stream){
  (void)out_size; (void)ws_size;
  static const int CH[8][2] = {{1,8},{8,16},{16,16},{16,16},{16,16},{16,32},{32,32},{32,32}};
  const float* x_in  = (const float*)d_in[0];
  const float* pos   = (const float*)d_in[1];
  const float* eattr = (const float*)d_in[2];
  const float *W[8], *Gm[8], *Bt[8], *Wfc;
  const int *ei, *batch;
  if (n_in >= 30){
    if (in_sizes[4] == 8){                 // interleaved: W0,g0,b0,W1,...
      int p = 3;
      for (int i = 0; i < 8; ++i){
        W[i]  = (const float*)d_in[p++];
        Gm[i] = (const float*)d_in[p++];
        Bt[i] = (const float*)d_in[p++];
      }
      Wfc = (const float*)d_in[p++];
      ei = (const int*)d_in[p++]; batch = (const int*)d_in[p++];
    } else {                               // key-sorted: W0..W7,Wfc,b0..b7,g0..g7
      for (int i = 0; i < 8; ++i) W[i]  = (const float*)d_in[3+i];
      Wfc = (const float*)d_in[11];
      for (int i = 0; i < 8; ++i) Bt[i] = (const float*)d_in[12+i];
      for (int i = 0; i < 8; ++i) Gm[i] = (const float*)d_in[20+i];
      ei = (const int*)d_in[28]; batch = (const int*)d_in[29];
    }
  } else {                                 // params flattened to one blob
    const float* blob = (const float*)d_in[3];
    size_t off = 0;
    for (int i = 0; i < 8; ++i){
      int ci = CH[i][0], co = CH[i][1];
      W[i]  = blob + off; off += (size_t)8*ci*co;
      Gm[i] = blob + off; off += co;
      Bt[i] = blob + off; off += co;
    }
    Wfc = blob + off;
    ei = (const int*)d_in[4]; batch = (const int*)d_in[5];
  }
  const int* esrc = ei;
  const int* edst = ei + EE_;

  // ---- workspace carve (bump allocator, 256B aligned) ----
  char* wp = (char*)d_ws;
  auto alloc = [&](size_t bytes)->char*{
    char* p = wp; wp += (bytes + 255) & ~(size_t)255; return p; };
  float* buf0  = (float*)alloc((size_t)NN_*32*4);
  float* buf1  = (float*)alloc((size_t)NN_*32*4);
  float* buf2  = (float*)alloc((size_t)NN_*32*4);
  float* agg   = (float*)alloc((size_t)NN_*32*4);
  float* stats = (float*)alloc(65*4);
  float* cnt1  = (float*)alloc((size_t)NN_*4);
  float* basis = (float*)alloc((size_t)EE_*8*4);       // reused for pooled basis
  int*   clus  = (int*)  alloc((size_t)NN_*4);
  float* cntc  = (float*)alloc((size_t)CC_*4);
  float* psum  = (float*)alloc((size_t)CC_*3*4);
  unsigned* xkey = (unsigned*)alloc((size_t)CC_*32*4);
  float* npos  = (float*)alloc((size_t)CC_*3*4);
  float* vmask = (float*)alloc((size_t)CC_*4);
  float* px0   = (float*)alloc((size_t)CC_*32*4);
  float* px1   = (float*)alloc((size_t)CC_*32*4);
  float* px2   = (float*)alloc((size_t)CC_*32*4);
  float* cnt2  = (float*)alloc((size_t)CC_*4);
  unsigned* mxkey = (unsigned*)alloc(256);
  unsigned* key16 = (unsigned*)alloc((size_t)G_*16*32*4);
  float* cnt16 = (float*)alloc((size_t)G_*16*4);
  float* feat  = (float*)alloc((size_t)G_*512*4);
  _Float16* Xh = (_Float16*)alloc((size_t)NN_*32*2);
  _Float16* Wh = (_Float16*)alloc((size_t)32*256*2);
  char* Yreg   = alloc((size_t)NN_*256*2);
  _Float16* Y  = (_Float16*)Yreg;
  // cart[E,3] aliases the tail of Y (pool-stage Y only needs CC_*256 f16 up front)
  float* cart  = (float*)(Yreg + (((size_t)CC_*256*2 + 255) & ~(size_t)255));

  auto cdiv = [](long long a, long long b){ return (int)((a + b - 1)/b); };

  // one conv block: GEMM (WMMA) -> edge scatter-mean -> ELU -> BN(+residual)
  auto run_conv = [&](const float* xin, int Cin, int Cout, float* xout,
                      const float* Wi, const float* gi, const float* bi,
                      int n, bool pool, const float* resid){
    int ncol = 8*Cout;
    k_prep_w<<<cdiv(32*ncol,256),256,0,stream>>>(Wi, Wh, Cin, Cout);
    k_pack_x<<<cdiv((long long)n*32,256),256,0,stream>>>(xin, Xh, n, Cin);
    dim3 gg(n/64, cdiv(ncol,128));
    k_gemm_wmma<<<gg,256,0,stream>>>(Xh, Wh, Y, n, ncol);
    k_zero<<<cdiv((long long)n*Cout,256),256,0,stream>>>(agg, (long long)n*Cout);
    k_zero<<<1,128,0,stream>>>(stats, 65);
    if (!pool)
      k_edge_msg<<<cdiv((long long)EE_*Cout,256),256,0,stream>>>(Y, basis, esrc, edst, agg, EE_, Cout);
    else
      k_edge_msg_pool<<<cdiv((long long)EE_*Cout,256),256,0,stream>>>(Y, basis, esrc, edst, clus, agg, EE_, Cout);
    const float* cnt  = pool ? cnt2  : cnt1;
    const float* nmsk = pool ? vmask : nullptr;
    k_elu_stats<<<cdiv((long long)n*Cout,256),256,0,stream>>>(agg, cnt, nmsk, stats, n, Cout);
    k_bn<<<cdiv((long long)n*Cout,256),256,0,stream>>>(agg, stats, gi, bi, resid, xout, n, Cout);
  };

  // ---- level-1 graph preprocessing (basis + in-degree), fixed for conv0..5 ----
  k_basis<<<cdiv(EE_,256),256,0,stream>>>(eattr, basis, EE_);
  k_zero<<<cdiv(NN_,256),256,0,stream>>>(cnt1, NN_);
  k_count<<<cdiv(EE_,256),256,0,stream>>>(edst, cnt1, EE_);

  // ---- conv blocks on full graph ----
  run_conv(x_in, 1,  8,  buf0, W[0], Gm[0], Bt[0], NN_, false, nullptr);
  run_conv(buf0, 8,  16, buf1, W[1], Gm[1], Bt[1], NN_, false, nullptr);
  run_conv(buf1, 16, 16, buf2, W[2], Gm[2], Bt[2], NN_, false, nullptr);  // sc
  run_conv(buf2, 16, 16, buf0, W[3], Gm[3], Bt[3], NN_, false, nullptr);
  run_conv(buf0, 16, 16, buf1, W[4], Gm[4], Bt[4], NN_, false, buf2);     // +sc
  run_conv(buf1, 16, 32, buf0, W[5], Gm[5], Bt[5], NN_, false, nullptr);

  // ---- voxel-grid max pool + new Cartesian edge attrs ----
  k_zero<<<cdiv(CC_,256),256,0,stream>>>(cntc, CC_);
  k_zero<<<cdiv((long long)CC_*3,256),256,0,stream>>>(psum, (long long)CC_*3);
  k_zero<<<cdiv((long long)CC_*32,256),256,0,stream>>>((float*)xkey, (long long)CC_*32);
  k_zero<<<cdiv(CC_,256),256,0,stream>>>(cnt2, CC_);
  k_zero<<<cdiv((long long)G_*16*32,256),256,0,stream>>>((float*)key16, (long long)G_*16*32);
  k_zero<<<1,256,0,stream>>>(cnt16, G_*16);
  k_pool_scatter<<<cdiv(NN_,256),256,0,stream>>>(buf0, pos, batch, clus, cntc, psum, xkey, NN_);
  k_pool_finalize<<<cdiv(CC_,256),256,0,stream>>>(cntc, psum, xkey, px0, npos, vmask, mxkey, CC_);
  k_pool_edge1<<<cdiv(EE_,256),256,0,stream>>>(esrc, edst, clus, npos, cart, mxkey, EE_);
  k_pool_attr_basis<<<cdiv(EE_,256),256,0,stream>>>(cart, mxkey, esrc, edst, clus, basis, cnt2, EE_);

  // ---- conv blocks on pooled graph (masked mean / masked BN) ----
  run_conv(px0, 32, 32, px1, W[6], Gm[6], Bt[6], CC_, true, nullptr);     // sc2 = px0
  run_conv(px1, 32, 32, px2, W[7], Gm[7], Bt[7], CC_, true, px0);         // +sc2

  // ---- 4x4 max pool -> [G,512] -> FC [G,2] ----
  k_maxpool16<<<cdiv(CC_,256),256,0,stream>>>(px2, npos, vmask, key16, cnt16, CC_);
  k_feat<<<cdiv((long long)G_*16*32,256),256,0,stream>>>(key16, cnt16, feat);
  k_fc<<<1,32,0,stream>>>(feat, Wfc, (float*)d_out);
}
